// RegionNonLocalBlock_17703855194547
// MI455X (gfx1250) — matrix operand order repaired and checked by reading, not compiled, not run on hardware
//
#include <hip/hip_runtime.h>
#include <hip/hip_bf16.h>

// MI455X (gfx1250) wave32 WMMA implementation of RegionNonLocalBlock.
// All GEMMs run through v_wmma_f32_16x16x32_bf16 (bf16 in, f32 accumulate).
// Workspace -> LDS staging uses the Tensor Data Mover (tensor_load_to_lds)
// with LDS row padding generated by the TDM pad engine.

typedef __attribute__((ext_vector_type(16))) __bf16        v16bf;
typedef __attribute__((ext_vector_type(8)))  float         v8f;
typedef __attribute__((ext_vector_type(4)))  unsigned int  u32x4;
typedef __attribute__((ext_vector_type(8)))  int           i32x8;
typedef __attribute__((ext_vector_type(4)))  int           i32x4;

union FragU { v16bf v; u32x4 u[2]; };

__device__ __forceinline__ unsigned short f2bf(float x) {
  unsigned u = __float_as_uint(x);
  u += 0x7FFFu + ((u >> 16) & 1u);       // round-to-nearest-even
  return (unsigned short)(u >> 16);
}

// A-operand (16x32 bf16): lane = M row; VGPR0-3 hold K {0..7}+8*hl, VGPR4-7 hold +16.
__device__ __forceinline__ v16bf ldsA(const unsigned short* buf, int stride,
                                      int row, int k0, int hl) {
  const unsigned short* p = buf + row * stride + k0 + 8 * hl;
  FragU f;
  f.u[0] = *(const u32x4*)(p);
  f.u[1] = *(const u32x4*)(p + 16);
  return f.v;
}
// B-operand (32x16 bf16): lane = N col; VGPR0-3 hold K {0..7}+8*hl, VGPR4-7 hold +16.
__device__ __forceinline__ v16bf ldsB(const unsigned short* buf, int stride,
                                      int col, int k0, int hl) {
  const unsigned short* p = buf + col * stride + k0 + 8 * hl;
  FragU f;
  f.u[0] = *(const u32x4*)(p);
  f.u[1] = *(const u32x4*)(p + 16);
  return f.v;
}

#define WMMA_BF16(acc, a, bmat)                                                 \
  acc = __builtin_amdgcn_wmma_f32_16x16x32_bf16(false, (a), false, (bmat),      \
                                                (short)0, (acc), false, false)

// ---------------------------------------------------------------------------
// TDM: load a 2D bf16 tile [dim1 rows][dim0 elems] from global to LDS,
// inserting pad_amount(+1) DWORDs of padding after every 2^(pad_interval+1)
// DWORDs (i.e. after each row), producing our padded LDS row strides.
//   D# group0: count=1 | lds_addr | global_addr(57b) | type=2
//   D# group1: data_size=1(2B), pad_enable, pad fields, dims, row stride
//   groups 2/3/4: zero (2D tensor, no iteration)
// ---------------------------------------------------------------------------
__device__ __forceinline__ void tdm_load_2d_bf16(unsigned lds_byte_off,
                                                 const void* gptr,
                                                 unsigned dim0, unsigned dim1,
                                                 unsigned pad_interval,
                                                 unsigned pad_amount) {
  unsigned long long ga = (unsigned long long)gptr;
  u32x4 g0;
  g0[0] = 1u;                                            // count=1, user D#
  g0[1] = lds_byte_off;                                  // lds_addr
  g0[2] = (unsigned)(ga & 0xFFFFFFFFu);                  // global_addr[31:0]
  g0[3] = (unsigned)((ga >> 32) & 0x01FFFFFFu) | (2u << 30);  // addr[56:32] | type=2
  i32x8 g1;
  g1[0] = (int)((1u << 16) |                             // data_size = 2 bytes
                (1u << 20) |                             // pad_enable
                (pad_interval << 22) | (pad_amount << 25));
  g1[1] = (int)(dim0 << 16);                             // tensor_dim0[15:0] @63:48
  g1[2] = (int)((dim0 >> 16) | (dim1 << 16));            // dim0 hi | tensor_dim1 lo
  g1[3] = (int)((dim1 >> 16) | (dim0 << 16));            // dim1 hi | tile_dim0 @127:112
  g1[4] = (int)dim1;                                     // tile_dim1 @143:128
  g1[5] = (int)dim0;                                     // tensor_dim0_stride lo
  g1[6] = 0;                                             // stride0 hi | stride1 lo
  g1[7] = 0;                                             // stride1 hi
  i32x4 z4 = {0, 0, 0, 0};
  i32x8 z8 = {0, 0, 0, 0, 0, 0, 0, 0};
  __builtin_amdgcn_tensor_load_to_lds(g0, g1, z4, z4, z8, 0);
}

// ---------------------------------------------------------------------------
// Pass 1: k/v = maxpool2(W @ cell + b), stored bf16 in workspace.
//   wsK layout [g][m=256][c=128]  (B-operand layout for the energy GEMM)
//   wsV layout [g][c=128][m=256]  (A-operand layout for the z GEMM)
// grid (256 cells, 2 {k,v}), 256 threads, LDS = 135168 B.
// ---------------------------------------------------------------------------
__global__ __launch_bounds__(256, 1) void kv_pool_kernel(
    const float* __restrict__ x,
    const float* __restrict__ Wk, const float* __restrict__ bk,
    const float* __restrict__ Wv, const float* __restrict__ bv,
    unsigned short* __restrict__ wsK, unsigned short* __restrict__ wsV) {
  extern __shared__ unsigned short smem[];
  unsigned short* Wl = smem;               // [128][264] bf16 (A layout, padded)
  unsigned short* Xl = smem + 128 * 264;   // [128][264] bf16 B; reused as f32 pool [128][132]
  float* Pl = (float*)Xl;

  const int g = blockIdx.x;
  const int sel = blockIdx.y;
  const int b = g >> 6, cell = g & 63, gr = cell >> 3, gc = cell & 7;
  const int tid = threadIdx.x;
  const int lane = tid & 31, l15 = lane & 15, hl = lane >> 4, wid = tid >> 5;

  const float* Wm   = sel ? Wv : Wk;
  const float* bias = sel ? bv : bk;

  for (int i = tid; i < 128 * 256; i += 256) {
    int row = i >> 8, k = i & 255;
    Wl[row * 264 + k] = f2bf(Wm[i]);
  }

  for (int t = 0; t < 8; ++t) {           // 8 chunks of 4 cell-rows (128 pixels)
    __syncthreads();
    // x chunk -> Xl[n][C] (coalesced 32-float row reads)
    for (int it = wid; it < 1024; it += 8) {
      int ch = it >> 2, lr = it & 3;
      int R = gr * 32 + 4 * t + lr;
      float v = x[(((b * 256 + ch) * 256) + R) * 256 + gc * 32 + lane];
      Xl[(lr * 32 + lane) * 264 + ch] = f2bf(v);
    }
    __syncthreads();

    // GEMM: [128 x 128] = W[128x256] @ x[256x128]; one 16-row strip per wave
    v8f acc[8] = {};
    const int m0 = wid * 16;
    for (int kt = 0; kt < 8; ++kt) {
      int k0 = kt * 32;
      v16bf a = ldsA(Wl, 264, m0 + l15, k0, hl);
      for (int nt = 0; nt < 8; ++nt) {
        v16bf bb = ldsB(Xl, 264, nt * 16 + l15, k0, hl);
        WMMA_BF16(acc[nt], a, bb);
      }
    }
    __syncthreads();                      // everyone done reading Xl
    for (int nt = 0; nt < 8; ++nt)
      for (int r = 0; r < 8; ++r)
        Pl[(m0 + r + 8 * hl) * 132 + nt * 16 + l15] = acc[nt][r];
    __syncthreads();

    // 2x2 maxpool + bias + bf16 store (chunk -> pooled rows 2t, 2t+1)
    for (int i = tid; i < 4096; i += 256) {
      int ch = i >> 5, pl = (i >> 4) & 1, pc = i & 15;
      const float* r0 = Pl + ch * 132 + (2 * pl) * 32 + 2 * pc;
      float m = fmaxf(fmaxf(r0[0], r0[1]), fmaxf(r0[32], r0[33])) + bias[ch];
      int mm = (2 * t + pl) * 16 + pc;
      if (sel == 0) wsK[((size_t)g * 256 + mm) * 128 + ch] = f2bf(m);
      else          wsV[((size_t)g * 128 + ch) * 256 + mm] = f2bf(m);
    }
  }
}

// ---------------------------------------------------------------------------
// Pass 2: fused q-conv + attention + output conv + residual.
// grid (8 n-tiles, 256 cells), 256 threads, LDS = 309248 B.
// ---------------------------------------------------------------------------
__global__ __launch_bounds__(256, 1) void attn_kernel(
    const float* __restrict__ x,
    const float* __restrict__ Wq, const float* __restrict__ bq,
    const float* __restrict__ Wo, const float* __restrict__ bo,
    const float* __restrict__ gamma,
    const unsigned short* __restrict__ wsK, const unsigned short* __restrict__ wsV,
    float* __restrict__ out) {
  extern __shared__ unsigned short smem[];
  unsigned short* Wl = smem;               // 34816 h: Wq [128][264] then Wo [256][136]
  unsigned short* KT = smem + 34816;       // 34816 h: k^T [m=256][c=128+8]   (B for energy)
  unsigned short* Vl = KT + 34816;         // 33792 h: v   [c=128][m=256+8]   (A for z)
  unsigned short* Xl = Vl + 33792;         // 33792 h: x-tile [n][C]; reused as attn [n][m]
  unsigned short* Ql = Xl + 33792;         // 17408 h: q^T [n][c]; reused as z [n][c]

  const int tileN = blockIdx.x;            // 0..7 -> pixels tileN*128 .. +127
  const int g = blockIdx.y;
  const int b = g >> 6, cell = g & 63, gr = cell >> 3, gc = cell & 7;
  const int tid = threadIdx.x;
  const int lane = tid & 31, l15 = lane & 15, hl = lane >> 4, wid = tid >> 5;

  // k^T and v: TDM async DMA into padded LDS layouts (one wave issues both).
  //   KT rows: 64 DWORDs data + 4 DWORDs pad  -> interval=5 (64 DW), amount=3 (4 DW)
  //   Vl rows: 128 DWORDs data + 4 DWORDs pad -> interval=6 (128 DW), amount=3
  if (wid == 0) {
    tdm_load_2d_bf16(34816u * 2u, wsK + (size_t)g * 32768, 128u, 256u, 5u, 3u);
    tdm_load_2d_bf16(69632u * 2u, wsV + (size_t)g * 32768, 256u, 128u, 6u, 3u);
  }

  // Wq -> A layout (overlaps with the TDM transfers)
  for (int i = tid; i < 128 * 256; i += 256) {
    int row = i >> 8, k = i & 255;
    Wl[row * 264 + k] = f2bf(Wq[i]);
  }
  // x tile -> Xl[n][C]
  for (int it = wid; it < 1024; it += 8) {
    int ch = it >> 2, lr = it & 3;
    int R = gr * 32 + 4 * tileN + lr;
    float v = x[(((b * 256 + ch) * 256) + R) * 256 + gc * 32 + lane];
    Xl[(lr * 32 + lane) * 264 + ch] = f2bf(v);
  }
  if (wid == 0) __builtin_amdgcn_s_wait_tensorcnt(0);
  __syncthreads();

  // q = Wq @ x_tile : M=c=128 (strip/wave), N=128, K=256; stage q^T (+bq) to Ql
  {
    v8f acc[8] = {};
    const int m0 = wid * 16;
    for (int kt = 0; kt < 8; ++kt) {
      int k0 = kt * 32;
      v16bf a = ldsA(Wl, 264, m0 + l15, k0, hl);
      for (int nt = 0; nt < 8; ++nt) {
        v16bf bb = ldsB(Xl, 264, nt * 16 + l15, k0, hl);
        WMMA_BF16(acc[nt], a, bb);
      }
    }
    for (int r = 0; r < 8; ++r) {
      int c = m0 + r + 8 * hl;
      float bqv = bq[c];
      for (int nt = 0; nt < 8; ++nt)
        Ql[(nt * 16 + l15) * 136 + c] = f2bf(acc[nt][r] + bqv);
    }
  }
  __syncthreads();

  // overlap: Wo -> Wl (read only in the final GEMM, after two more barriers)
  for (int i = tid; i < 256 * 128; i += 256) {
    int row = i >> 7, k = i & 127;
    Wl[row * 136 + k] = f2bf(Wo[i]);
  }

  // energy = q^T k : M=n (strip/wave), N=m=256 (16 tiles), K=c=128
  v8f e[16] = {};
  {
    const int m0 = wid * 16;
    for (int kt = 0; kt < 4; ++kt) {
      int k0 = kt * 32;
      v16bf a = ldsA(Ql, 136, m0 + l15, k0, hl);
      for (int mt = 0; mt < 16; ++mt) {
        v16bf bb = ldsB(KT, 136, mt * 16 + l15, k0, hl);
        WMMA_BF16(e[mt], a, bb);
      }
    }
    // softmax over m (256): VGPR r holds row (m0 + r + 8*hl), cols = mt*16 + l15.
    // width-16 xor shuffles reduce across the 16 lanes of each half-wave.
    for (int r = 0; r < 8; ++r) {
      float mx = -3.0e38f;
      for (int mt = 0; mt < 16; ++mt) mx = fmaxf(mx, e[mt][r]);
      for (int o = 1; o < 16; o <<= 1) mx = fmaxf(mx, __shfl_xor(mx, o, 16));
      float s = 0.f;
      for (int mt = 0; mt < 16; ++mt) {
        float p = __expf(e[mt][r] - mx);
        e[mt][r] = p;
        s += p;
      }
      for (int o = 1; o < 16; o <<= 1) s += __shfl_xor(s, o, 16);
      float inv = 1.0f / s;
      int nrow = m0 + r + 8 * hl;
      for (int mt = 0; mt < 16; ++mt)
        Xl[nrow * 264 + mt * 16 + l15] = f2bf(e[mt][r] * inv);  // attn[n][m]
    }
  }
  __syncthreads();

  // z = v @ attn^T : M=c (strip/wave), N=128, K=m=256; stage z^T [n][c] to Ql
  {
    v8f acc[8] = {};
    const int m0 = wid * 16;
    for (int kt = 0; kt < 8; ++kt) {
      int k0 = kt * 32;
      v16bf a = ldsA(Vl, 264, m0 + l15, k0, hl);
      for (int nt = 0; nt < 8; ++nt) {
        v16bf bb = ldsB(Xl, 264, nt * 16 + l15, k0, hl);
        WMMA_BF16(acc[nt], a, bb);
      }
    }
    for (int r = 0; r < 8; ++r) {
      int c = m0 + r + 8 * hl;
      for (int nt = 0; nt < 8; ++nt)
        Ql[(nt * 16 + l15) * 136 + c] = f2bf(acc[nt][r]);
    }
  }
  __syncthreads();

  // out = gamma * (Wo @ z + bo) + x : M=C=256 (2 strips/wave), N=128, K=c=128
  const float g0 = gamma[0];
  for (int sb = 0; sb < 2; ++sb) {
    const int m0 = (wid + sb * 8) * 16;
    v8f acc[8] = {};
    for (int kt = 0; kt < 4; ++kt) {
      int k0 = kt * 32;
      v16bf a = ldsA(Wl, 136, m0 + l15, k0, hl);
      for (int nt = 0; nt < 8; ++nt) {
        v16bf bb = ldsB(Ql, 136, nt * 16 + l15, k0, hl);
        WMMA_BF16(acc[nt], a, bb);
      }
    }
    for (int r = 0; r < 8; ++r) {
      int ch = m0 + r + 8 * hl;
      float bov = bo[ch];
      for (int nt = 0; nt < 8; ++nt) {
        int n = tileN * 128 + nt * 16 + l15;
        int rc = n >> 5, colc = n & 31;
        int addr = (((b * 256 + ch) * 256) + gr * 32 + rc) * 256 + gc * 32 + colc;
        out[addr] = g0 * (acc[nt][r] + bov) + x[addr];   // residual
      }
    }
  }
}

// ---------------------------------------------------------------------------
extern "C" void kernel_launch(void* const* d_in, const int* in_sizes, int n_in,
                              void* d_out, int out_size, void* d_ws, size_t ws_size,
                              hipStream_t stream) {
  (void)in_sizes; (void)n_in; (void)out_size; (void)ws_size;
  const float* x     = (const float*)d_in[0];
  const float* Wq    = (const float*)d_in[1];
  const float* bq    = (const float*)d_in[2];
  const float* Wk    = (const float*)d_in[3];
  const float* bk    = (const float*)d_in[4];
  const float* Wv    = (const float*)d_in[5];
  const float* bv    = (const float*)d_in[6];
  const float* Wo    = (const float*)d_in[7];
  const float* bo    = (const float*)d_in[8];
  const float* gamma = (const float*)d_in[9];
  float* out = (float*)d_out;

  unsigned short* wsK = (unsigned short*)d_ws;                 // [256][256][128] bf16
  unsigned short* wsV = wsK + (size_t)256 * 256 * 128;         // [256][128][256] bf16

  const int lds1 = 2 * (128 * 264 + 128 * 264);                          // 135168 B
  const int lds2 = 2 * (34816 + 34816 + 33792 + 33792 + 17408);          // 309248 B
  (void)hipFuncSetAttribute((const void*)kv_pool_kernel,
                            hipFuncAttributeMaxDynamicSharedMemorySize, lds1);
  (void)hipFuncSetAttribute((const void*)attn_kernel,
                            hipFuncAttributeMaxDynamicSharedMemorySize, lds2);

  kv_pool_kernel<<<dim3(256, 2), 256, lds1, stream>>>(x, Wk, bk, Wv, bv, wsK, wsV);
  attn_kernel<<<dim3(8, 256), 256, lds2, stream>>>(x, Wq, bq, Wo, bo, gamma,
                                                   wsK, wsV, out);
}
